// SamplerFig8GATCoordinateTime_48155173323058
// MI455X (gfx1250) — matrix operand order repaired and checked by reading, not compiled, not run on hardware
//
#include <hip/hip_runtime.h>
#include <hip/hip_bf16.h>

#define NN 8192
#define NPART 8
#define GCOLS 48     // padded B columns: d values + ones col + zero pad
#define JCHUNK 512   // j's staged in LDS per chunk (2 chunks per partition)

typedef __attribute__((ext_vector_type(16))) _Float16 v16h;
typedef __attribute__((ext_vector_type(8)))  _Float16 v8h;
typedef __attribute__((ext_vector_type(8)))  float    v8f;
typedef __attribute__((ext_vector_type(4)))  float    v4f;

__device__ __forceinline__ float lrelu02(float x) { return fmaxf(x, 0.2f * x); }

// ---------------- Kernel 0: pack adjacency (fp32 [N,N,1]) into bitmask ----
__global__ __launch_bounds__(256) void pack_mask(const float* __restrict__ adj,
                                                 unsigned* __restrict__ maskbits) {
    const int w = blockIdx.x * 256 + threadIdx.x;      // one 32-bit word per thread
    const float* p = adj + (size_t)w * 32;
    unsigned bits = 0;
#pragma unroll
    for (int k = 0; k < 32; ++k)
        bits |= (p[k] > 0.5f ? 1u : 0u) << k;
    maskbits[w] = bits;
}

// ---------------- Projection: g = X @ W, scores, transposed f16 values ----
template <int DIN, int DOUT>
__global__ __launch_bounds__(256) void gat_proj(const float* __restrict__ X,
                                                const float* __restrict__ W,
                                                const float* __restrict__ a,
                                                _Float16* __restrict__ gT16,
                                                float* __restrict__ g32,
                                                float* __restrict__ s_src,
                                                float* __restrict__ s_dst) {
    const int i = blockIdx.x * 256 + threadIdx.x;
    if (i >= NN) return;
    float xr[DIN];
#pragma unroll
    for (int k = 0; k < DIN; ++k) xr[k] = X[(size_t)i * DIN + k];
    float ssrc = 0.f, sdst = 0.f;
#pragma unroll
    for (int c = 0; c < DOUT; ++c) {
        float acc = 0.f;
#pragma unroll
        for (int k = 0; k < DIN; ++k) acc = fmaf(xr[k], W[k * DOUT + c], acc);
        gT16[(size_t)c * NN + i] = (_Float16)acc;
        if (g32) g32[(size_t)i * DOUT + c] = acc;
        ssrc = fmaf(acc, a[c], ssrc);
        sdst = fmaf(acc, a[DOUT + c], sdst);
    }
    gT16[(size_t)DOUT * NN + i] = (_Float16)1.0f;     // ones column -> row sum Z
#pragma unroll
    for (int c = DOUT + 1; c < GCOLS; ++c) gT16[(size_t)c * NN + i] = (_Float16)0.0f;
    s_src[i] = ssrc;
    s_dst[i] = sdst;
}

// ---------------- Global max of s_dst (softmax shift upper bound) ---------
__global__ __launch_bounds__(256) void reduce_max(const float* __restrict__ s,
                                                  float* __restrict__ smax) {
    __shared__ float sm[256];
    float m = -1e30f;
    for (int j = threadIdx.x; j < NN; j += 256) m = fmaxf(m, s[j]);
    sm[threadIdx.x] = m;
    __syncthreads();
    for (int st = 128; st > 0; st >>= 1) {
        if (threadIdx.x < st) sm[threadIdx.x] = fmaxf(sm[threadIdx.x], sm[threadIdx.x + st]);
        __syncthreads();
    }
    if (threadIdx.x == 0) *smax = sm[0];
}

// ---------------- WMMA attention: partial (P @ [g | 1]) per j-partition ---
// Block = 4 waves, one 16-row tile each, SAME j-partition: B panel staged in LDS.
__global__ __launch_bounds__(128) void gat_attn(const float* __restrict__ s_src,
                                                const float* __restrict__ s_dst,
                                                const float* __restrict__ smax_p,
                                                const unsigned* __restrict__ maskbits,
                                                const _Float16* __restrict__ gT16,
                                                float* __restrict__ partials) {
    __shared__ _Float16 gLds[GCOLS * JCHUNK];   // 48 KB: B panel slice, [col][JCHUNK]
    __shared__ float    sdLds[JCHUNK];          // 2 KB: s_dst slice

    const int lane  = threadIdx.x & 31;
    const int itile = blockIdx.x * 4 + (threadIdx.x >> 5);   // 0..511 (16-row tile)
    const int part  = blockIdx.y;                            // 0..NPART-1
    const int m  = lane & 15;                                // A row / B-C col
    const int hi = lane >> 4;                                // K-half select
    const int row = itile * 16 + m;

    const float smax = *smax_p;
    const float ss   = s_src[row];
    const float mrow = lrelu02(ss + smax);                   // valid per-row max bound

    v8f c0 = {}, c1 = {}, c2 = {};
    const int jbeg = part * (NN / NPART);

    for (int chunk = 0; chunk < (NN / NPART) / JCHUNK; ++chunk) {
        const int cbase = jbeg + chunk * JCHUNK;

        // ---- cooperative stage: global -> LDS (uniform trip counts) ----
#pragma unroll 4
        for (int idx = threadIdx.x; idx < GCOLS * (JCHUNK / 8); idx += 128) {
            const int c  = idx / (JCHUNK / 8);
            const int jo = (idx % (JCHUNK / 8)) * 8;
            *(v8h*)(gLds + c * JCHUNK + jo) =
                *(const v8h*)(gT16 + (size_t)c * NN + cbase + jo);
        }
#pragma unroll
        for (int j = threadIdx.x; j < JCHUNK; j += 128) sdLds[j] = s_dst[cbase + j];
        __syncthreads();

        for (int jl = 0; jl < JCHUNK; jl += 32) {
            const unsigned mw = maskbits[(size_t)row * (NN / 32) + ((cbase + jl) >> 5)];
            const float* sa = sdLds + jl + hi * 8;
            const v4f sA0 = *(const v4f*)(sa);
            const v4f sA1 = *(const v4f*)(sa + 4);
            const v4f sB0 = *(const v4f*)(sa + 16);
            const v4f sB1 = *(const v4f*)(sa + 20);

            // B tiles (32x16 f16) from LDS: lane = col n, K = hi*8 + t (t<8), 16 + hi*8 + t-8 (t>=8)
            const int boff = jl + hi * 8;
            const v16h b0 = __builtin_shufflevector(*(const v8h*)(gLds + (m)      * JCHUNK + boff), *(const v8h*)(gLds + (m)      * JCHUNK + boff + 16), 0,1,2,3,4,5,6,7,8,9,10,11,12,13,14,15);
            const v16h b1 = __builtin_shufflevector(*(const v8h*)(gLds + (16 + m) * JCHUNK + boff), *(const v8h*)(gLds + (16 + m) * JCHUNK + boff + 16), 0,1,2,3,4,5,6,7,8,9,10,11,12,13,14,15);
            const v16h b2 = __builtin_shufflevector(*(const v8h*)(gLds + (32 + m) * JCHUNK + boff), *(const v8h*)(gLds + (32 + m) * JCHUNK + boff + 16), 0,1,2,3,4,5,6,7,8,9,10,11,12,13,14,15);

            float sd[16];
#pragma unroll
            for (int t = 0; t < 4; ++t) { sd[t] = sA0[t]; sd[4 + t] = sA1[t];
                                          sd[8 + t] = sB0[t]; sd[12 + t] = sB1[t]; }

            // A tile (16x32 f16), ISA lane layout:
            // halves 0..7  -> K = hi*8 + t        (mask bit hi*8 + t)
            // halves 8..15 -> K = 16 + hi*8 + t-8 (mask bit 8 + hi*8 + t)
            v16h aa;
#pragma unroll
            for (int t = 0; t < 16; ++t) {
                const float e = lrelu02(ss + sd[t]);
                float w = __expf(e - mrow);                  // unconditional: no branch
                const int bitpos = (t < 8) ? (hi * 8 + t) : (8 + hi * 8 + t);
                w = ((mw >> bitpos) & 1u) ? w : 0.f;         // v_cndmask
                aa[t] = (_Float16)w;
            }

            c0 = __builtin_amdgcn_wmma_f32_16x16x32_f16(false, aa, false, b0, (short)0, c0, false, false);
            c1 = __builtin_amdgcn_wmma_f32_16x16x32_f16(false, aa, false, b1, (short)0, c1, false, false);
            c2 = __builtin_amdgcn_wmma_f32_16x16x32_f16(false, aa, false, b2, (short)0, c2, false, false);
        }
        __syncthreads();   // all waves done reading before restage
    }

    // C layout: VGPR r -> row hi*8+r, col = m. Store partials[part][row][48].
#pragma unroll
    for (int r = 0; r < 8; ++r) {
        const int rm = itile * 16 + hi * 8 + r;
        const size_t base = ((size_t)part * NN + rm) * GCOLS + m;
        partials[base + 0]  = c0[r];
        partials[base + 16] = c1[r];
        partials[base + 32] = c2[r];
    }
}

// ---------------- Combine partitions: /Z then ELU --------------------------
template <int D>
__global__ __launch_bounds__(256) void gat_combine(const float* __restrict__ partials,
                                                   float* __restrict__ hout) {
    const int i = blockIdx.x * 256 + threadIdx.x;
    if (i >= NN) return;
    float acc[D + 1];
#pragma unroll
    for (int c = 0; c <= D; ++c) acc[c] = 0.f;
    for (int p = 0; p < NPART; ++p) {
        const float* src = partials + ((size_t)p * NN + i) * GCOLS;
#pragma unroll
        for (int c = 0; c <= D; ++c) acc[c] += src[c];
    }
    const float Z = acc[D];
    const float inv = Z > 0.f ? 1.f / Z : 0.f;
#pragma unroll
    for (int c = 0; c < D; ++c) {
        const float v  = acc[c] * inv;
        const float ev = __expf(v) - 1.f;                   // unconditional
        hout[(size_t)i * D + c] = v > 0.f ? v : ev;         // ELU, branchless
    }
}

// ---------------- Layer-3 attention (row 0 only) fused with MLP -----------
__global__ __launch_bounds__(256) void gat_row0_mlp(const float* __restrict__ s_src,
                                                    const float* __restrict__ s_dst,
                                                    const float* __restrict__ smax_p,
                                                    const unsigned* __restrict__ maskbits,
                                                    const float* __restrict__ g32,
                                                    const float* __restrict__ Wm1,
                                                    const float* __restrict__ bm1,
                                                    const float* __restrict__ Wm2,
                                                    const float* __restrict__ bm2,
                                                    float* __restrict__ out) {
    __shared__ float lds[256 * 34];
    __shared__ float f[34];
    __shared__ float hid[128];

    const float ss = s_src[0];
    const float m0 = lrelu02(ss + *smax_p);

    float acc[34];
#pragma unroll
    for (int c = 0; c < 34; ++c) acc[c] = 0.f;

    for (int j = threadIdx.x; j < NN; j += 256) {
        const float e = lrelu02(ss + s_dst[j]);
        float w = __expf(e - m0);                             // unconditional
        w = ((maskbits[j >> 5] >> (j & 31)) & 1u) ? w : 0.f;  // row 0 of mask
        const float* g = g32 + (size_t)j * 33;
#pragma unroll
        for (int c = 0; c < 33; ++c) acc[c] = fmaf(w, g[c], acc[c]);
        acc[33] += w;
    }
#pragma unroll
    for (int c = 0; c < 34; ++c) lds[threadIdx.x * 34 + c] = acc[c];
    __syncthreads();

    if (threadIdx.x < 34) {
        float s = 0.f;
        for (int t = 0; t < 256; ++t) s += lds[t * 34 + threadIdx.x];
        f[threadIdx.x] = s;
    }
    __syncthreads();
    if (threadIdx.x < 33) {
        const float Z = f[33];
        f[threadIdx.x] = Z > 0.f ? f[threadIdx.x] / Z : 0.f;  // h3[0][c], no ELU
    }
    __syncthreads();

    if (threadIdx.x < 128) {
        float s = bm1[threadIdx.x];
#pragma unroll
        for (int k = 0; k < 33; ++k) s = fmaf(f[k], Wm1[k * 128 + threadIdx.x], s);
        hid[threadIdx.x] = fmaxf(s, 0.01f * s);               // leaky_relu 0.01
    }
    __syncthreads();
    if (threadIdx.x < 27) {
        float s = bm2[threadIdx.x];
        for (int k = 0; k < 128; ++k) s = fmaf(hid[k], Wm2[k * 27 + threadIdx.x], s);
        out[threadIdx.x] = s;
    }
}

// --------------------------------------------------------------------------
extern "C" void kernel_launch(void* const* d_in, const int* in_sizes, int n_in,
                              void* d_out, int out_size, void* d_ws, size_t ws_size,
                              hipStream_t stream) {
    const float* x   = (const float*)d_in[0];
    const float* adj = (const float*)d_in[1];
    const float* W1  = (const float*)d_in[2];
    const float* a1  = (const float*)d_in[3];
    const float* W2  = (const float*)d_in[4];
    const float* a2  = (const float*)d_in[5];
    const float* W3  = (const float*)d_in[6];
    const float* a3  = (const float*)d_in[7];
    const float* Wm1 = (const float*)d_in[8];
    const float* bm1 = (const float*)d_in[9];
    const float* Wm2 = (const float*)d_in[10];
    const float* bm2 = (const float*)d_in[11];
    float* out = (float*)d_out;

    char* ws = (char*)d_ws;
    size_t off = 0;
    auto alloc = [&](size_t bytes) -> void* {
        void* p = ws + off;
        off = (off + bytes + 255) & ~(size_t)255;
        return p;
    };
    unsigned* maskbits = (unsigned*)alloc((size_t)NN * (NN / 32) * 4);   // 8 MB
    float*    s_src    = (float*)alloc((size_t)NN * 4);
    float*    s_dst    = (float*)alloc((size_t)NN * 4);
    float*    smax     = (float*)alloc(256);
    _Float16* gT16     = (_Float16*)alloc((size_t)GCOLS * NN * 2);       // 768 KB
    float*    g32      = (float*)alloc((size_t)NN * 33 * 4);
    float*    partials = (float*)alloc((size_t)NPART * NN * GCOLS * 4);  // 12.6 MB
    float*    hbuf     = (float*)alloc((size_t)NN * 32 * 4);
    (void)ws_size; (void)in_sizes; (void)n_in; (void)out_size;

    const dim3 attnGrid(512 / 4, NPART);

    // Pack adjacency once: 256 MB -> 8 MB bitmask
    pack_mask<<<(NN * (NN / 32)) / 256, 256, 0, stream>>>(adj, maskbits);

    // ---- Layer 1: x[8192,33] -> h[8192,32] (ELU) ----
    gat_proj<33, 32><<<NN / 256, 256, 0, stream>>>(x, W1, a1, gT16, nullptr, s_src, s_dst);
    reduce_max<<<1, 256, 0, stream>>>(s_dst, smax);
    gat_attn<<<attnGrid, 128, 0, stream>>>(s_src, s_dst, smax, maskbits, gT16, partials);
    gat_combine<32><<<NN / 256, 256, 0, stream>>>(partials, hbuf);

    // ---- Layer 2: h[8192,32] -> h[8192,32] (ELU) ----
    gat_proj<32, 32><<<NN / 256, 256, 0, stream>>>(hbuf, W2, a2, gT16, nullptr, s_src, s_dst);
    reduce_max<<<1, 256, 0, stream>>>(s_dst, smax);
    gat_attn<<<attnGrid, 128, 0, stream>>>(s_src, s_dst, smax, maskbits, gT16, partials);
    gat_combine<32><<<NN / 256, 256, 0, stream>>>(partials, hbuf);

    // ---- Layer 3 (values for all nodes, attention row 0 only) + MLP ----
    gat_proj<32, 33><<<NN / 256, 256, 0, stream>>>(hbuf, W3, a3, gT16, g32, s_src, s_dst);
    reduce_max<<<1, 256, 0, stream>>>(s_dst, smax);
    gat_row0_mlp<<<1, 256, 0, stream>>>(s_src, s_dst, smax, maskbits, g32,
                                        Wm1, bm1, Wm2, bm2, out);
}
